// GCRN_66718021976734
// MI455X (gfx1250) — matrix operand order repaired and checked by reading, not compiled, not run on hardware
//
#include <hip/hip_runtime.h>
#include <hip/hip_bf16.h>

namespace {
constexpr int Bc = 32, Nc = 512, Ec = 4, Fc = 128, Gc = 64;
constexpr int EMBc = 128, HIDc = 256, NEDGE = 8192;
constexpr float EPSc = 1e-5f;
constexpr int EG = Ec * Gc;                     // 256
constexpr long long Mrows = (long long)Bc * Nc; // 16384
}

typedef __attribute__((ext_vector_type(16))) _Float16 v16h;
typedef __attribute__((ext_vector_type(8)))  _Float16 v8h;
typedef __attribute__((ext_vector_type(8)))  float    v8f;

__device__ __forceinline__ int lane_id() { return (int)(threadIdx.x & 31u); }

__device__ __forceinline__ v8f wmma16x16x32(v16h a, v16h b, v8f c) {
  return __builtin_amdgcn_wmma_f32_16x16x32_f16(false, a, false, b, (short)0, c, false, false);
}

// A fragment (16x32 f16, ISA 7.12.2): lane holds row (lane&15); halves 0..7 = K base..base+7,
// halves 8..15 = K base+16..base+23, base = (lane&16)?8:0. Two contiguous 16B runs -> 2x b128.
__device__ __forceinline__ v16h load_afrag_f16(const _Float16* __restrict__ s, int ld, int row0, int k0) {
  const int lane = lane_id();
  const _Float16* p = s + (long long)(row0 + (lane & 15)) * ld + k0 + ((lane & 16) ? 8 : 0);
  v8h lo = *(const v8h*)(p);
  v8h hi = *(const v8h*)(p + 16);
  return __builtin_shufflevector(lo, hi, 0, 1, 2, 3, 4, 5, 6, 7, 8, 9, 10, 11, 12, 13, 14, 15);
}

// B fragment (32x16 f16) from K-major (transposed) storage s[col][k], leading dim ldk.
// Lane holds col (lane&15); halves 0..7 = K base..base+7, halves 8..15 = K base+16..base+23, base = k0 + ((lane&16)?8:0). 2x b128.
__device__ __forceinline__ v16h load_bfrag_f16t(const _Float16* __restrict__ s, int ldk, int k0, int col0) {
  const int lane = lane_id();
  const _Float16* p = s + (long long)(col0 + (lane & 15)) * ldk + k0 + ((lane & 16) ? 8 : 0);
  v8h lo = *(const v8h*)(p);
  v8h hi = *(const v8h*)(p + 16);
  return __builtin_shufflevector(lo, hi, 0, 1, 2, 3, 4, 5, 6, 7, 8, 9, 10, 11, 12, 13, 14, 15);
}

// ---------------- pre-pass kernels ----------------

__global__ void fill_zero_k(float* __restrict__ p, long long n) {
  long long i = (long long)blockIdx.x * blockDim.x + threadIdx.x;
  long long stride = (long long)gridDim.x * blockDim.x;
  for (; i < n; i += stride) p[i] = 0.f;
}

__global__ void cvt_f16_k(const float* __restrict__ src, _Float16* __restrict__ dst, long long n) {
  long long i = (long long)blockIdx.x * blockDim.x + threadIdx.x;
  if (i < n) dst[i] = (_Float16)src[i];
}

// batched transpose+convert: dst[b][c][r] = src[b][r][c]
__global__ void btranspose_cvt_k(const float* __restrict__ src, _Float16* __restrict__ dst,
                                 int rows, int cols, long long total) {
  long long i = (long long)blockIdx.x * blockDim.x + threadIdx.x;
  if (i >= total) return;
  int c = (int)(i % cols);
  long long t = i / cols;
  int r = (int)(t % rows);
  long long bidx = t / rows;
  dst[(bidx * cols + c) * (long long)rows + r] = (_Float16)src[i];
}

__global__ void build_adj_k(const long long* __restrict__ edges, float* __restrict__ Adj) {
  long long idx = (long long)blockIdx.x * blockDim.x + threadIdx.x; // over B*E*NEDGE
  if (idx >= (long long)Bc * Ec * NEDGE) return;
  int i = (int)(idx % NEDGE);
  int be = (int)(idx / NEDGE);
  const long long* ep = edges + (long long)be * 2 * NEDGE;
  int src = (int)ep[i];
  int dst = (int)ep[NEDGE + i];
  atomicAdd(&Adj[((long long)be * Nc + src) * Nc + dst], 1.0f);
}

// ---------------- GEMM stages ----------------

// Wh = Xh[b] (N x F) @ Ws[e] (F x G). One wave per 16-row strip, 4 accumulators (16x64).
// Outputs: Wh32 row-major [n][g] (for attn vectors), Wh16t transposed [g][n] (B operand of attn GEMM).
__global__ void __launch_bounds__(32) wh_gemm_k(const _Float16* __restrict__ Xh,
                                                const _Float16* __restrict__ Wst,
                                                float* __restrict__ Wh32,
                                                _Float16* __restrict__ Wh16t) {
  const int be = blockIdx.x, bb = be / Ec, ee = be % Ec;
  const int row0 = blockIdx.y * 16;
  const _Float16* Xb = Xh + (long long)bb * Nc * Fc;
  const _Float16* Wt = Wst + (long long)ee * Gc * Fc;   // [g][f]
  v8f acc0 = {}, acc1 = {}, acc2 = {}, acc3 = {};
#pragma unroll
  for (int k0 = 0; k0 < Fc; k0 += 32) {
    v16h a  = load_afrag_f16(Xb, Fc, row0, k0);
    v16h b0 = load_bfrag_f16t(Wt, Fc, k0, 0);
    v16h b1 = load_bfrag_f16t(Wt, Fc, k0, 16);
    v16h b2 = load_bfrag_f16t(Wt, Fc, k0, 32);
    v16h b3 = load_bfrag_f16t(Wt, Fc, k0, 48);
    acc0 = wmma16x16x32(a, b0, acc0);
    acc1 = wmma16x16x32(a, b1, acc1);
    acc2 = wmma16x16x32(a, b2, acc2);
    acc3 = wmma16x16x32(a, b3, acc3);
  }
  const int lane = lane_id();
  const int cl = lane & 15;
  const int rbase = row0 + ((lane & 16) ? 8 : 0);
  float* o32 = Wh32 + (long long)be * Nc * Gc;
  _Float16* o16t = Wh16t + (long long)be * Gc * Nc;
  const v8f* accs[4] = {&acc0, &acc1, &acc2, &acc3};
#pragma unroll
  for (int t = 0; t < 4; ++t) {
    const int col = t * 16 + cl;
#pragma unroll
    for (int j = 0; j < 8; ++j) {
      float v = (*accs[t])[j];
      o32[(long long)(rbase + j) * Gc + col] = v;
      o16t[(long long)col * Nc + (rbase + j)] = (_Float16)v;
    }
  }
}

// Wh1[b,e,n] = sum_g Wh*a[e,g]; Wh2 with a[e,G+g]
__global__ void attn_vec_k(const float* __restrict__ Wh32, const float* __restrict__ av,
                           float* __restrict__ Wh1, float* __restrict__ Wh2) {
  long long idx = (long long)blockIdx.x * blockDim.x + threadIdx.x; // over B*E*N
  if (idx >= (long long)Bc * Ec * Nc) return;
  int be = (int)(idx / Nc);
  int ee = be % Ec;
  const float* w = Wh32 + idx * Gc;
  const float* a1 = av + (long long)ee * 2 * Gc;
  float s1 = 0.f, s2 = 0.f;
#pragma unroll 8
  for (int g = 0; g < Gc; ++g) { float v = w[g]; s1 += v * a1[g]; s2 += v * a1[Gc + g]; }
  Wh1[idx] = s1; Wh2[idx] = s2;
}

// fused: scores -> softmax -> P = att*A (f16 in LDS) -> Htile = P @ Wh (WMMA, B from transposed f16)
__global__ void __launch_bounds__(128) attn_h_k(const float* __restrict__ Adj,
                                                const float* __restrict__ Wh1,
                                                const float* __restrict__ Wh2,
                                                const _Float16* __restrict__ Wh16t,
                                                _Float16* __restrict__ Hflat) {
  __shared__ float wh2s[Nc];
  __shared__ float rowscale[16];
  __shared__ _Float16 p16[16 * Nc];
  const int be = blockIdx.x, bb = be / Ec, ee = be % Ec;
  const int n0 = blockIdx.y * 16;
  const int tid = (int)threadIdx.x;
  for (int i = tid; i < Nc; i += 128) wh2s[i] = Wh2[(long long)be * Nc + i];
  __syncthreads();
  // 8 threads per row, 16 rows
  const int row = tid >> 3, sub = tid & 7;
  const float* Arow = Adj + ((long long)be * Nc + n0 + row) * Nc;
  const float wh1v = Wh1[(long long)be * Nc + n0 + row];
  float mx = -3.4e38f;
  for (int c = sub; c < Nc; c += 8) {
    float s = (wh1v + wh2s[c]) * Arow[c];
    mx = fmaxf(mx, s);
  }
#pragma unroll
  for (int m = 1; m < 8; m <<= 1) mx = fmaxf(mx, __shfl_xor(mx, m, 32));
  float sum = 0.f;
  for (int c = sub; c < Nc; c += 8) {
    float a = Arow[c];
    float s = (wh1v + wh2s[c]) * a;
    float ev = __expf(s - mx);
    sum += ev;                       // dense softmax: zeros contribute exp(-mx)
    p16[row * Nc + c] = (_Float16)(ev * a);
  }
#pragma unroll
  for (int m = 1; m < 8; m <<= 1) sum += __shfl_xor(sum, m, 32);
  if (sub == 0) rowscale[row] = 1.f / sum;
  __syncthreads();
  // GEMM: (16 x 512) @ (512 x 64), 4 waves x one 16x16 tile.
  // Two independent accumulator chains over even/odd K-chunks to break WMMA RAW serialization.
  const int wv = tid >> 5;
  const _Float16* WbeT = Wh16t + (long long)be * Gc * Nc;   // [g][m]
  v8f accE = {}, accO = {};
#pragma unroll 4
  for (int k0 = 0; k0 < Nc; k0 += 64) {
    v16h aE = load_afrag_f16(p16, Nc, 0, k0);
    v16h bE = load_bfrag_f16t(WbeT, Nc, k0, wv * 16);
    v16h aO = load_afrag_f16(p16, Nc, 0, k0 + 32);
    v16h bO = load_bfrag_f16t(WbeT, Nc, k0 + 32, wv * 16);
    accE = wmma16x16x32(aE, bE, accE);
    accO = wmma16x16x32(aO, bO, accO);
  }
  v8f acc = accE + accO;
  const int lane = lane_id();
  const int colg = ee * Gc + wv * 16 + (lane & 15);
  const int rb = (lane & 16) ? 8 : 0;
#pragma unroll
  for (int j = 0; j < 8; ++j) {
    long long grow = (long long)bb * Nc + n0 + rb + j;
    Hflat[grow * EG + colg] = (_Float16)(acc[j] * rowscale[rb + j]);
  }
}

// h = Hflat (16384 x 256) @ lin0_w (256 x 256) + b ; BN sums. 4 waves, each a 16x64 strip.
__global__ void __launch_bounds__(128) lin0_gemm_k(const _Float16* __restrict__ Hf,
                                                   const _Float16* __restrict__ Wt, // [col][k]
                                                   const float* __restrict__ bias,
                                                   float* __restrict__ hout,
                                                   float* __restrict__ bnsum,
                                                   float* __restrict__ bnsq) {
  const int row0 = blockIdx.x * 16;
  const int wv = (int)(threadIdx.x >> 5);
  const int colb = wv * 64;
  v8f acc0 = {}, acc1 = {}, acc2 = {}, acc3 = {};
#pragma unroll
  for (int k0 = 0; k0 < EG; k0 += 32) {
    v16h a  = load_afrag_f16(Hf, EG, row0, k0);
    v16h b0 = load_bfrag_f16t(Wt, EG, k0, colb);
    v16h b1 = load_bfrag_f16t(Wt, EG, k0, colb + 16);
    v16h b2 = load_bfrag_f16t(Wt, EG, k0, colb + 32);
    v16h b3 = load_bfrag_f16t(Wt, EG, k0, colb + 48);
    acc0 = wmma16x16x32(a, b0, acc0);
    acc1 = wmma16x16x32(a, b1, acc1);
    acc2 = wmma16x16x32(a, b2, acc2);
    acc3 = wmma16x16x32(a, b3, acc3);
  }
  const int lane = lane_id();
  const int cl = lane & 15;
  const int rbase = row0 + ((lane & 16) ? 8 : 0);
  const v8f* accs[4] = {&acc0, &acc1, &acc2, &acc3};
#pragma unroll
  for (int t = 0; t < 4; ++t) {
    const int col = colb + t * 16 + cl;
    const float bv = bias[col];
    float s = 0.f, s2 = 0.f;
#pragma unroll
    for (int j = 0; j < 8; ++j) {
      float v = (*accs[t])[j] + bv;
      hout[(long long)(rbase + j) * HIDc + col] = v;
      s += v; s2 += v * v;
    }
    atomicAdd(&bnsum[col], s);
    atomicAdd(&bnsq[col], s2);
  }
}

__global__ void bn_final_k(const float* __restrict__ bnsum, const float* __restrict__ bnsq,
                           const float* __restrict__ gamma, const float* __restrict__ beta,
                           float* __restrict__ scale, float* __restrict__ shift) {
  int c = (int)(blockIdx.x * blockDim.x + threadIdx.x);
  if (c >= HIDc) return;
  const float inv = 1.f / (float)(Bc * Nc);
  float mu = bnsum[c] * inv;
  float var = bnsq[c] * inv - mu * mu;
  float rs = rsqrtf(var + EPSc) * gamma[c];
  scale[c] = rs;
  shift[c] = beta[c] - mu * rs;
}

__global__ void elu_act_k(const float* __restrict__ h, const float* __restrict__ scale,
                          const float* __restrict__ shift, _Float16* __restrict__ act) {
  long long i = (long long)blockIdx.x * blockDim.x + threadIdx.x;
  if (i >= Mrows * HIDc) return;
  int c = (int)(i % HIDc);
  float v = h[i] * scale[c] + shift[c];
  v = v > 0.f ? v : (__expf(v) - 1.f);
  act[i] = (_Float16)v;
}

// out = act (16384 x 256) @ lin1_w (256 x 128) + b. 2 waves, each a 16x64 strip.
__global__ void __launch_bounds__(64) lin1_gemm_k(const _Float16* __restrict__ act,
                                                  const _Float16* __restrict__ Wt, // [col][k]
                                                  const float* __restrict__ bias,
                                                  float* __restrict__ out) {
  const int row0 = blockIdx.x * 16;
  const int wv = (int)(threadIdx.x >> 5);
  const int colb = wv * 64;
  v8f acc0 = {}, acc1 = {}, acc2 = {}, acc3 = {};
#pragma unroll
  for (int k0 = 0; k0 < HIDc; k0 += 32) {
    v16h a  = load_afrag_f16(act, HIDc, row0, k0);
    v16h b0 = load_bfrag_f16t(Wt, HIDc, k0, colb);
    v16h b1 = load_bfrag_f16t(Wt, HIDc, k0, colb + 16);
    v16h b2 = load_bfrag_f16t(Wt, HIDc, k0, colb + 32);
    v16h b3 = load_bfrag_f16t(Wt, HIDc, k0, colb + 48);
    acc0 = wmma16x16x32(a, b0, acc0);
    acc1 = wmma16x16x32(a, b1, acc1);
    acc2 = wmma16x16x32(a, b2, acc2);
    acc3 = wmma16x16x32(a, b3, acc3);
  }
  const int lane = lane_id();
  const int cl = lane & 15;
  const int rbase = row0 + ((lane & 16) ? 8 : 0);
  const v8f* accs[4] = {&acc0, &acc1, &acc2, &acc3};
#pragma unroll
  for (int t = 0; t < 4; ++t) {
    const int col = colb + t * 16 + cl;
    const float bv = bias[col];
#pragma unroll
    for (int j = 0; j < 8; ++j)
      out[(long long)(rbase + j) * EMBc + col] = (*accs[t])[j] + bv;
  }
}

extern "C" void kernel_launch(void* const* d_in, const int* in_sizes, int n_in,
                              void* d_out, int out_size, void* d_ws, size_t ws_size,
                              hipStream_t stream) {
  const float*      X        = (const float*)d_in[0];
  const long long*  edges    = (const long long*)d_in[1];
  const float*      Ws       = (const float*)d_in[2];
  const float*      av       = (const float*)d_in[3];
  const float*      lin0_w   = (const float*)d_in[4];
  const float*      lin0_b   = (const float*)d_in[5];
  const float*      bn_gamma = (const float*)d_in[6];
  const float*      bn_beta  = (const float*)d_in[7];
  const float*      lin1_w   = (const float*)d_in[8];
  const float*      lin1_b   = (const float*)d_in[9];
  float* out = (float*)d_out;
  char* ws = (char*)d_ws;

  size_t off = 0;
  float*    Adj   = (float*)(ws + off);    off += (size_t)Bc * Ec * Nc * Nc * 4;   // 134 MB
  float*    Wh32  = (float*)(ws + off);    off += (size_t)Bc * Ec * Nc * Gc * 4;   // 16.8 MB
  _Float16* Wh16t = (_Float16*)(ws + off); off += (size_t)Bc * Ec * Nc * Gc * 2;   // 8.4 MB
  float*    Wh1   = (float*)(ws + off);    off += (size_t)Bc * Ec * Nc * 4;
  float*    Wh2   = (float*)(ws + off);    off += (size_t)Bc * Ec * Nc * 4;
  _Float16* Hf    = (_Float16*)(ws + off); off += (size_t)Mrows * EG * 2;          // 8.4 MB
  float*    h32   = (float*)(ws + off);    off += (size_t)Mrows * HIDc * 4;        // 16.8 MB
  _Float16* act   = (_Float16*)(ws + off); off += (size_t)Mrows * HIDc * 2;        // 8.4 MB
  _Float16* Xh    = (_Float16*)(ws + off); off += (size_t)Bc * Nc * Fc * 2;        // 4.2 MB
  _Float16* Wst   = (_Float16*)(ws + off); off += (size_t)Ec * Fc * Gc * 2;        // [e][g][f]
  _Float16* l0t   = (_Float16*)(ws + off); off += (size_t)EG * HIDc * 2;           // [col][k]
  _Float16* l1t   = (_Float16*)(ws + off); off += (size_t)HIDc * EMBc * 2;         // [col][k]
  float*    bnsum  = (float*)(ws + off);   off += HIDc * 4;
  float*    bnsq   = (float*)(ws + off);   off += HIDc * 4;
  float*    bscale = (float*)(ws + off);   off += HIDc * 4;
  float*    bshift = (float*)(ws + off);   off += HIDc * 4;

  // pre-pass: zero + operand conversion/transposition
  fill_zero_k<<<4096, 256, 0, stream>>>(Adj, (long long)Bc * Ec * Nc * Nc);
  fill_zero_k<<<2, 256, 0, stream>>>(bnsum, (long long)(2 * HIDc)); // bnsum+bnsq contiguous
  cvt_f16_k<<<(int)((Bc * Nc * Fc) / 256), 256, 0, stream>>>(X, Xh, (long long)Bc * Nc * Fc);
  btranspose_cvt_k<<<(Ec * Fc * Gc) / 256, 256, 0, stream>>>(Ws, Wst, Fc, Gc, (long long)Ec * Fc * Gc);
  btranspose_cvt_k<<<(EG * HIDc) / 256, 256, 0, stream>>>(lin0_w, l0t, EG, HIDc, (long long)EG * HIDc);
  btranspose_cvt_k<<<(HIDc * EMBc) / 256, 256, 0, stream>>>(lin1_w, l1t, HIDc, EMBc, (long long)HIDc * EMBc);
  build_adj_k<<<(Bc * Ec * NEDGE) / 256, 256, 0, stream>>>(edges, Adj);

  // main pipeline
  wh_gemm_k<<<dim3(Bc * Ec, Nc / 16), 32, 0, stream>>>(Xh, Wst, Wh32, Wh16t);
  attn_vec_k<<<(Bc * Ec * Nc) / 256, 256, 0, stream>>>(Wh32, av, Wh1, Wh2);
  attn_h_k<<<dim3(Bc * Ec, Nc / 16), 128, 0, stream>>>(Adj, Wh1, Wh2, Wh16t, Hf);
  lin0_gemm_k<<<dim3((int)(Mrows / 16)), 128, 0, stream>>>(Hf, l0t, lin0_b, h32, bnsum, bnsq);
  bn_final_k<<<1, HIDc, 0, stream>>>(bnsum, bnsq, bn_gamma, bn_beta, bscale, bshift);
  elu_act_k<<<(int)((Mrows * HIDc) / 256), 256, 0, stream>>>(h32, bscale, bshift, act);
  lin1_gemm_k<<<(int)(Mrows / 16), 64, 0, stream>>>(act, l1t, lin1_b, out);
}
